// MultiDomainPLEFENDModel_38147899523618
// MI455X (gfx1250) — compile-verified, matrix operand order checked by reading, not run on hardware
//
#include <hip/hip_runtime.h>
#include <hip/hip_bf16.h>
#include <math.h>

typedef __attribute__((ext_vector_type(16))) _Float16 v16h;
typedef __attribute__((ext_vector_type(8)))  _Float16 v8h;
typedef __attribute__((ext_vector_type(8)))  float    v8f;

#define DEVI static __device__ __forceinline__

constexpr int cB = 8, cL = 96, cD = 768, cF = 320, cP = 12, cNE = 24, cNC = 1536, cE = 18;

DEVI int ixmap(int e, int dm) { return e < 6 ? dm * 6 + e : e + 6; }

DEVI float act_apply(float x, int act) {
  switch (act) {
    case 1: return x / (1.f + __expf(-x));                                  // SiLU
    case 2: return x > 0.f ? x : 0.f;                                       // ReLU
    case 3: return 0.5f * x * (1.f + erff(x * 0.70710678118654752440f));    // exact GELU
    default: return x;
  }
}

// ---------------------------------------------------------------- utilities
__global__ void k_zero(float* p, long n) {
  long t = (long)blockIdx.x * blockDim.x + threadIdx.x;
  if (t < n) p[t] = 0.f;
}

__global__ void k_f32_to_f16(const float* __restrict__ s, _Float16* __restrict__ d, long n) {
  long t = (long)blockIdx.x * blockDim.x + threadIdx.x;
  if (t < n) d[t] = (_Float16)s[t];
}

// W[o, d, j] (OIH, f32)  ->  BT[o, j*768 + d] (f16)
__global__ void k_transpose_w(const float* __restrict__ W, _Float16* __restrict__ BT,
                              int kw, long n) {
  long t = (long)blockIdx.x * blockDim.x + threadIdx.x;
  if (t >= n) return;
  int kd = cD * kw;
  long o = t / kd;
  int r = (int)(t - o * kd);
  int d = r / kw, j = r - d * kw;
  BT[o * kd + j * cD + d] = (_Float16)W[t];
}

// ------------------------------------------------------- WMMA implicit-GEMM conv
DEVI v16h cat_frag(v8h lo, v8h hi) {
  v16h r;
#pragma unroll
  for (int i = 0; i < 8; ++i) { r[i] = lo[i]; r[8 + i] = hi[i]; }
  return r;
}

// C[m,n] = sum_k A[m,k] * BT[n,k];  m=(b,p) over output positions, n=channel.
// Block = 8 waves: M-supertile 64 rows (shared A staged in LDS), each wave one
// 16-col N-tile with 4 accumulators -> B fragment reused 4x per K-step.
// Epilogue: out[b, e, kidx*64+c] = max_p relu(C + bias[n]) via uint atomicMax.
__global__ void __launch_bounds__(256)
k_conv_wmma(const _Float16* __restrict__ X, const _Float16* __restrict__ BT,
            const float* __restrict__ bias, float* __restrict__ out, int kw, int kidx) {
  __shared__ _Float16 As[64 * 32];  // 64 rows x 32 K-halfs per step (4 KB)

  const int K = kw * cD;
  const int Lout = cL - kw + 1;
  const int M = cB * Lout;
  const int lane = threadIdx.x & 31;
  const int wave = threadIdx.x >> 5;
  const int msuper = blockIdx.x;          // 64-row M supertile
  const int ntile = blockIdx.y * 8 + wave;
  const int ml = lane & 15;               // row-in-tile (A) / col-in-tile (B,C)
  const int hh = lane >> 4;               // K-half selector

  // per-thread A-fill assignment: one b128 chunk of the 64x32 tile
  const int rf = threadIdx.x >> 2, cf = threadIdx.x & 3;
  int amf = msuper * 64 + rf;
  if (amf >= M) amf = M - 1;              // clamp; OOB rows masked in epilogue
  const int bf = amf / Lout, pf = amf - bf * Lout;
  const _Float16* growf = X + ((long)bf * cL + pf) * cD + cf * 8;  // im2col row

  const int col = ntile * 16 + ml;
  const _Float16* brow = BT + (long)col * K;

  v8f acc[4] = {{}, {}, {}, {}};
  for (int kk = 0; kk < K; kk += 32) {
    __syncthreads();
    *(v8h*)&As[rf * 32 + cf * 8] = *(const v8h*)(growf + kk);
    if (kk + 512 < K) __builtin_prefetch((const void*)(brow + kk + 512), 0, 0);
    v16h bfrag = cat_frag(*(const v8h*)(brow + kk + hh * 8),
                          *(const v8h*)(brow + kk + 16 + hh * 8));
    __syncthreads();
#pragma unroll
    for (int mi = 0; mi < 4; ++mi) {
      const _Float16* ar = &As[(mi * 16 + ml) * 32 + hh * 8];
      v16h afrag = cat_frag(*(const v8h*)ar, *(const v8h*)(ar + 16));
      acc[mi] = __builtin_amdgcn_wmma_f32_16x16x32_f16(false, afrag, false, bfrag,
                                                       (short)0, acc[mi], false, false);
    }
  }

  const int e = col >> 6, c = col & 63;
  const float bv = bias[col];
#pragma unroll
  for (int mi = 0; mi < 4; ++mi) {
#pragma unroll
    for (int r = 0; r < 8; ++r) {
      int m = msuper * 64 + mi * 16 + r + 8 * hh;
      if (m < M) {
        float v = acc[mi][r] + bv;
        if (v < 0.f) v = 0.f;
        int b = m / Lout;
        atomicMax((unsigned int*)&out[((long)b * cNE + e) * cF + kidx * 64 + c],
                  __float_as_uint(v));
      }
    }
  }
}

// ---------------------------------------------------------------- small f32 ops
// Y[m,n] = act(sum_k X[m,k]*W[k,n] + bias[n])
__global__ void k_dense_linear(const float* __restrict__ X, const float* __restrict__ W,
                               const float* __restrict__ Bv, float* __restrict__ Y,
                               int M, int K, int N, int act) {
  long t = (long)blockIdx.x * blockDim.x + threadIdx.x;
  if (t >= (long)M * N) return;
  int n = (int)(t % N), m = (int)(t / N);
  const float* x = X + (long)m * K;
  const float* w = W + n;
  float s = Bv ? Bv[n] : 0.f;
  for (int k = 0; k < K; ++k) s += x[k] * w[(long)k * N];
  Y[t] = act_apply(s, act);
}

// Batched over 18 experts with ix-mapped params; X shared (xes=0) or per-expert.
// y_binner=1 writes Y[(m*18+e)*N+n] (B-major), else Y[((e*M)+m)*N+n].
__global__ void k_expert_linear(const float* __restrict__ X, long xes,
                                const float* __restrict__ W, const float* __restrict__ Bv,
                                float* __restrict__ Y, int M, int K, int N, int act,
                                int dm, int y_binner) {
  long t = (long)blockIdx.x * blockDim.x + threadIdx.x;
  if (t >= (long)cE * M * N) return;
  int n = (int)(t % N);
  long r = t / N;
  int m = (int)(r % M), e = (int)(r / M);
  int xe = ixmap(e, dm);
  const float* x = X + e * xes + (long)m * K;
  const float* w = W + (long)xe * K * N + n;
  float s = Bv ? Bv[(long)xe * N + n] : 0.f;
  for (int k = 0; k < K; ++k) s += x[k] * w[(long)k * N];
  long yi = y_binner ? ((long)m * cE + e) * N + n : ((long)e * M + m) * N + n;
  Y[yi] = act_apply(s, act);
}

__global__ void k_softmax_rows(float* __restrict__ X, int M, int N) {
  int r = blockIdx.x * blockDim.x + threadIdx.x;
  if (r >= M) return;
  float* x = X + (long)r * N;
  float mx = -1e30f;
  for (int n = 0; n < N; ++n) mx = fmaxf(mx, x[n]);
  float s = 0.f;
  for (int n = 0; n < N; ++n) { float e = __expf(x[n] - mx); x[n] = e; s += e; }
  float inv = 1.f / s;
  for (int n = 0; n < N; ++n) x[n] *= inv;
}

__global__ void k_weighted_pool(const float* __restrict__ X, const float* __restrict__ A,
                                float* __restrict__ Y) {  // Y[b,d]=sum_l A[b,l]X[b,l,d]
  int t = blockIdx.x * blockDim.x + threadIdx.x;
  if (t >= cB * cD) return;
  int d = t % cD, b = t / cD;
  float s = 0.f;
  for (int l = 0; l < cL; ++l) s += A[b * cL + l] * X[((long)b * cL + l) * cD + d];
  Y[t] = s;
}

// Y[b,f] = sum_e G[b,e] * T[b, col(e), f];  T layout [B, Estride, F]
__global__ void k_mix_experts(const float* __restrict__ G, const float* __restrict__ T,
                              float* __restrict__ Y, int Estride, int use_ix, int dm) {
  int t = blockIdx.x * blockDim.x + threadIdx.x;
  if (t >= cB * cF) return;
  int f = t % cF, b = t / cF;
  float s = 0.f;
  for (int e = 0; e < cE; ++e) {
    int col = use_ix ? ixmap(e, dm) : e;
    s += G[b * cE + e] * T[((long)b * Estride + col) * cF + f];
  }
  Y[t] = s;
}

// final[b,f] = sum_e G[b,e] * blk[e,b,f]
__global__ void k_mix_ebf(const float* __restrict__ G, const float* __restrict__ blk,
                          float* __restrict__ Y) {
  int t = blockIdx.x * blockDim.x + threadIdx.x;
  if (t >= cB * cF) return;
  int f = t % cF, b = t / cF;
  float s = 0.f;
  for (int e = 0; e < cE; ++e) s += G[b * cE + e] * blk[((long)e * cB + b) * cF + f];
  Y[t] = s;
}

__global__ void k_scale_add(const float* A, const float* B, float* Y, float sc, long n) {
  long t = (long)blockIdx.x * blockDim.x + threadIdx.x;
  if (t < n) Y[t] = sc * (A[t] + B[t]);
}

__global__ void k_pack3(const float* __restrict__ a, const float* __restrict__ b,
                        const float* __restrict__ c, float* __restrict__ tok) {
  int t = blockIdx.x * blockDim.x + threadIdx.x;
  if (t >= cB * 3 * cF) return;
  int f = t % cF, s = (t / cF) % 3, bb = t / (3 * cF);
  const float* src = s == 0 ? a : (s == 1 ? b : c);
  tok[t] = src[bb * cF + f];
}

__global__ void k_bcast_tokens(const float* __restrict__ tok, float* __restrict__ xb) {
  long t = (long)blockIdx.x * blockDim.x + threadIdx.x;
  if (t < (long)cE * cB * 3 * cF) xb[t] = tok[t % (cB * 3 * cF)];
}

__global__ void k_layernorm_expert(const float* __restrict__ X, const float* __restrict__ G,
                                   const float* __restrict__ Bt, float* __restrict__ Y,
                                   int rows, int dm) {
  int r = blockIdx.x * blockDim.x + threadIdx.x;
  if (r >= cE * rows) return;
  int e = r / rows, xe = ixmap(e, dm);
  const float* x = X + (long)r * cF;
  float* y = Y + (long)r * cF;
  float m = 0.f;
  for (int f = 0; f < cF; ++f) m += x[f];
  m /= cF;
  float va = 0.f;
  for (int f = 0; f < cF; ++f) { float d = x[f] - m; va += d * d; }
  va /= cF;
  float rinv = rsqrtf(va + 1e-6f);
  for (int f = 0; f < cF; ++f)
    y[f] = (x[f] - m) * rinv * G[(long)xe * cF + f] + Bt[(long)xe * cF + f];
}

// qkv [E,B,3,960]: inner = j*320 + h*40 + d -> attout [E,B,3,320]
__global__ void k_attention(const float* __restrict__ qkv, float* __restrict__ o) {
  int t = blockIdx.x * blockDim.x + threadIdx.x;
  if (t >= cE * cB * 8) return;
  int h = t & 7, b = (t >> 3) & 7, e = t >> 6;
  const float* base = qkv + ((long)e * 24 + b * 3) * 960 + h * 40;
  const float sc = 0.15811388300841897f;  // 1/sqrt(40)
  float s[3][3];
#pragma unroll
  for (int ti = 0; ti < 3; ++ti) {
#pragma unroll
    for (int tj = 0; tj < 3; ++tj) {
      float a = 0.f;
      for (int d = 0; d < 40; ++d) a += base[ti * 960 + d] * base[tj * 960 + 320 + d];
      s[ti][tj] = a * sc;
    }
    float mx = fmaxf(s[ti][0], fmaxf(s[ti][1], s[ti][2]));
    float sm = 0.f;
#pragma unroll
    for (int tj = 0; tj < 3; ++tj) { s[ti][tj] = __expf(s[ti][tj] - mx); sm += s[ti][tj]; }
    float inv = 1.f / sm;
#pragma unroll
    for (int tj = 0; tj < 3; ++tj) s[ti][tj] *= inv;
  }
  float* ob = o + ((long)e * 24 + b * 3) * cF + h * 40;
#pragma unroll
  for (int ti = 0; ti < 3; ++ti)
    for (int d = 0; d < 40; ++d) {
      float a = 0.f;
#pragma unroll
      for (int tj = 0; tj < 3; ++tj) a += s[ti][tj] * base[tj * 960 + 640 + d];
      ob[ti * cF + d] = a;
    }
}

__global__ void k_residual_add(float* __restrict__ X, const float* __restrict__ D, long n) {
  long t = (long)blockIdx.x * blockDim.x + threadIdx.x;
  if (t < n) X[t] += D[t];
}

__global__ void k_mean_tokens(const float* __restrict__ X, float* __restrict__ blk) {
  int t = blockIdx.x * blockDim.x + threadIdx.x;
  if (t >= cE * cB * cF) return;
  int f = t % cF, b = (t / cF) % cB, e = t / (cF * cB);
  float s = 0.f;
  for (int tt = 0; tt < 3; ++tt) s += X[(((long)e * cB + b) * 3 + tt) * cF + f];
  blk[t] = s * (1.f / 3.f);
}

__global__ void k_proto_scores(const float* __restrict__ X, const float* __restrict__ Pw,
                               float* __restrict__ S) {
  int t = blockIdx.x * blockDim.x + threadIdx.x;
  if (t >= cB * cP) return;
  int p = t % cP, b = t / cP;
  float s = 0.f;
  for (int f = 0; f < cF; ++f) s += X[b * cF + f] * Pw[p * cF + f];
  S[t] = s * 0.05590169943749474f;  // 1/sqrt(320)
}

__global__ void k_sinkhorn(const float* __restrict__ st, const float* __restrict__ si,
                           float* __restrict__ T) {
  int b = blockIdx.x * blockDim.x + threadIdx.x;
  if (b >= cB) return;
  float C[cP][cP];
  float mx = -1e30f;
  for (int i = 0; i < cP; ++i)
    for (int j = 0; j < cP; ++j) {
      C[i][j] = fabsf(st[b * cP + i] - si[b * cP + j]);
      mx = fmaxf(mx, C[i][j]);
    }
  float inv = 1.f / (mx + 1e-8f);
  for (int i = 0; i < cP; ++i)
    for (int j = 0; j < cP; ++j) C[i][j] *= inv;
  const float eps = 0.05f, ieps = 20.f;
  const float logm = -logf((float)cP);
  float u[cP], v[cP], un[cP], vn[cP];
  for (int i = 0; i < cP; ++i) { u[i] = 0.f; v[i] = 0.f; }
  for (int it = 0; it < 20; ++it) {
    for (int i = 0; i < cP; ++i) {
      float m2 = -1e30f;
      for (int j = 0; j < cP; ++j) m2 = fmaxf(m2, -C[i][j] * ieps + v[j]);
      float s = 0.f;
      for (int j = 0; j < cP; ++j) s += expf(-C[i][j] * ieps + v[j] - m2);
      un[i] = eps * (logm - (m2 + logf(s))) + v[i];
    }
    for (int i = 0; i < cP; ++i) u[i] = un[i];
    for (int j = 0; j < cP; ++j) {
      float m2 = -1e30f;
      for (int i = 0; i < cP; ++i) m2 = fmaxf(m2, -C[i][j] * ieps + u[i]);
      float s = 0.f;
      for (int i = 0; i < cP; ++i) s += expf(-C[i][j] * ieps + u[i] - m2);
      vn[j] = eps * (logm - (m2 + logf(s))) + u[j];
    }
    for (int j = 0; j < cP; ++j) v[j] = vn[j];
  }
  for (int i = 0; i < cP; ++i)
    for (int j = 0; j < cP; ++j)
      T[b * cP * cP + i * cP + j] = expf((-C[i][j] + u[i] + v[j]) * ieps);
}

__global__ void k_concat2(const float* __restrict__ A, const float* __restrict__ B,
                          float* __restrict__ Y) {  // [B,320]+[B,320] -> [B,640]
  int t = blockIdx.x * blockDim.x + threadIdx.x;
  if (t >= cB * 2 * cF) return;
  int c = t % (2 * cF), b = t / (2 * cF);
  Y[t] = c < cF ? A[b * cF + c] : B[b * cF + c - cF];
}

__global__ void k_sigmoid_out(const float* __restrict__ lg, float* __restrict__ out, int dm) {
  int b = blockIdx.x * blockDim.x + threadIdx.x;
  if (b < cB) out[b * 2 + dm] = 1.f / (1.f + __expf(-lg[b]));
}

// ================================================================= launch
static inline unsigned gb(long n) { return (unsigned)((n + 255) / 256); }

extern "C" void kernel_launch(void* const* d_in, const int* in_sizes, int n_in,
                              void* d_out, int out_size, void* d_ws, size_t ws_size,
                              hipStream_t stream) {
  (void)in_sizes; (void)n_in; (void)out_size; (void)ws_size;
  const float* IN[80];
  for (int i = 0; i < n_in && i < 80; ++i) IN[i] = (const float*)d_in[i];
  float* out = (float*)d_out;

  // ---- workspace bump allocator (256B aligned)
  char* wp = (char*)d_ws;
  auto alloc = [&](size_t bytes) -> void* {
    void* p = (void*)wp;
    wp += (bytes + 255) & ~(size_t)255;
    return p;
  };
  _Float16* xt16 = (_Float16*)alloc((size_t)cB * cL * cD * 2);
  _Float16* xi16 = (_Float16*)alloc((size_t)cB * cL * cD * 2);
  _Float16* wtr  = (_Float16*)alloc((size_t)cNC * cD * 10 * 2);  // reused per conv
  float* te = (float*)alloc((size_t)cB * cNE * cF * 4);
  float* ie = (float*)alloc((size_t)cB * cNE * cF * 4);
  float* scores = (float*)alloc((size_t)cB * cL * 4);
  float* tp = (float*)alloc((size_t)cB * cD * 4);
  float* ip = (float*)alloc((size_t)cB * cD * 4);
  float* gh768 = (float*)alloc((size_t)cB * cD * 4);
  float* gt = (float*)alloc((size_t)cB * cE * 4);
  float* gi = (float*)alloc((size_t)cB * cE * 4);
  float* tout = (float*)alloc((size_t)cB * cF * 4);
  float* iout = (float*)alloc((size_t)cB * cF * 4);
  float* fin  = (float*)alloc((size_t)cB * cF * 4);
  float* gh160 = (float*)alloc((size_t)cB * 160 * 4);
  float* gf = (float*)alloc((size_t)cB * cE * 4);
  float* feh = (float*)alloc((size_t)cE * cB * cF * 4);
  float* feBEF = (float*)alloc((size_t)cB * cE * cF * 4);
  float* fout = (float*)alloc((size_t)cB * cF * 4);
  float* tokens = (float*)alloc((size_t)cB * 3 * cF * 4);
  float* xbuf = (float*)alloc((size_t)cE * 24 * cF * 4);
  float* hbuf = (float*)alloc((size_t)cE * 24 * cF * 4);
  float* qkvb = (float*)alloc((size_t)cE * 24 * 960 * 4);
  float* attb = (float*)alloc((size_t)cE * 24 * cF * 4);
  float* projb = (float*)alloc((size_t)cE * 24 * cF * 4);
  float* mlpb = (float*)alloc((size_t)cE * 24 * 1280 * 4);
  float* blk = (float*)alloc((size_t)cE * cB * cF * 4);
  float* nh1 = (float*)alloc((size_t)cB * cF * 4);
  float* nh2 = (float*)alloc((size_t)cB * 160 * 4);
  float* gn = (float*)alloc((size_t)cB * cE * 4);
  float* finl = (float*)alloc((size_t)cB * cF * 4);
  float* st = (float*)alloc((size_t)cB * cP * 4);
  float* si = (float*)alloc((size_t)cB * cP * 4);
  float* Tm = (float*)alloc((size_t)cB * cP * cP * 4);
  float* dh = (float*)alloc((size_t)cB * 128 * 4);
  float* dis = (float*)alloc((size_t)cB * cF * 4);
  float* cat = (float*)alloc((size_t)cB * 2 * cF * 4);
  float* fh = (float*)alloc((size_t)cB * cF * 4);
  float* fused = (float*)alloc((size_t)cB * cF * 4);
  float* ch = (float*)alloc((size_t)cB * 384 * 4);
  float* lgt = (float*)alloc((size_t)cB * 4);

  const long nx = (long)cB * cL * cD;
  k_f32_to_f16<<<gb(nx), 256, 0, stream>>>(IN[0], xt16, nx);
  k_f32_to_f16<<<gb(nx), 256, 0, stream>>>(IN[1], xi16, nx);
  k_zero<<<gb((long)cB * cNE * cF), 256, 0, stream>>>(te, (long)cB * cNE * cF);
  k_zero<<<gb((long)cB * cNE * cF), 256, 0, stream>>>(ie, (long)cB * cNE * cF);

  // --------- TextCNN expert banks (dominant compute) via WMMA implicit GEMM
  const int KSa[5] = {1, 2, 3, 5, 10};
  for (int m = 0; m < 2; ++m) {
    const _Float16* x16 = m == 0 ? xt16 : xi16;
    float* dst = m == 0 ? te : ie;
    int base = m == 0 ? 2 : 14;
    for (int i = 0; i < 5; ++i) {
      int kw = KSa[i];
      long nw = (long)cNC * cD * kw;
      k_transpose_w<<<gb(nw), 256, 0, stream>>>(IN[base + 2 * i], wtr, kw, nw);
      int Lout = cL - kw + 1, Mtot = cB * Lout;
      dim3 g((Mtot + 63) / 64, cNC / 16 / 8);   // 64-row M supertile per block
      k_conv_wmma<<<g, 256, 0, stream>>>(x16, wtr, IN[base + 2 * i + 1], dst, kw, i);
    }
  }

  // --------- attention pooling per modality
  k_dense_linear<<<gb(cB * cL), 256, 0, stream>>>(IN[0], IN[12], IN[13], scores,
                                                  cB * cL, cD, 1, 0);
  k_softmax_rows<<<1, 32, 0, stream>>>(scores, cB, cL);
  k_weighted_pool<<<gb(cB * cD), 256, 0, stream>>>(IN[0], scores, tp);
  k_dense_linear<<<gb(cB * cL), 256, 0, stream>>>(IN[1], IN[24], IN[25], scores,
                                                  cB * cL, cD, 1, 0);
  k_softmax_rows<<<1, 32, 0, stream>>>(scores, cB, cL);
  k_weighted_pool<<<gb(cB * cD), 256, 0, stream>>>(IN[1], scores, ip);

  for (int dm = 0; dm < 2; ++dm) {
    // modality gates
    k_dense_linear<<<gb(cB * cD), 256, 0, stream>>>(tp, IN[26] + (long)dm * cD * cD,
                                                    IN[27] + dm * cD, gh768, cB, cD, cD, 1);
    k_dense_linear<<<gb(cB * cE), 256, 0, stream>>>(gh768, IN[28] + (long)dm * cD * cE,
                                                    IN[29] + dm * cE, gt, cB, cD, cE, 0);
    k_softmax_rows<<<1, 32, 0, stream>>>(gt, cB, cE);
    k_dense_linear<<<gb(cB * cD), 256, 0, stream>>>(ip, IN[30] + (long)dm * cD * cD,
                                                    IN[31] + dm * cD, gh768, cB, cD, cD, 1);
    k_dense_linear<<<gb(cB * cE), 256, 0, stream>>>(gh768, IN[32] + (long)dm * cD * cE,
                                                    IN[33] + dm * cE, gi, cB, cD, cE, 0);
    k_softmax_rows<<<1, 32, 0, stream>>>(gi, cB, cE);
    k_mix_experts<<<gb(cB * cF), 256, 0, stream>>>(gt, te, tout, cNE, 1, dm);
    k_mix_experts<<<gb(cB * cF), 256, 0, stream>>>(gi, ie, iout, cNE, 1, dm);
    k_scale_add<<<gb(cB * cF), 256, 0, stream>>>(tout, iout, fin, 0.5f, cB * cF);

    // feature experts + feature gate
    k_expert_linear<<<gb((long)cE * cB * cF), 256, 0, stream>>>(
        fin, 0L, IN[44], IN[45], feh, cB, cF, cF, 1, dm, 0);
    k_expert_linear<<<gb((long)cE * cB * cF), 256, 0, stream>>>(
        feh, (long)cB * cF, IN[46], IN[47], feBEF, cB, cF, cF, 0, dm, 1);
    k_dense_linear<<<gb(cB * 160), 256, 0, stream>>>(fin, IN[34] + (long)dm * cF * 160,
                                                     IN[35] + dm * 160, gh160, cB, cF, 160, 1);
    k_dense_linear<<<gb(cB * cE), 256, 0, stream>>>(gh160, IN[36] + (long)dm * 160 * cE,
                                                    IN[37] + dm * cE, gf, cB, 160, cE, 0);
    k_softmax_rows<<<1, 32, 0, stream>>>(gf, cB, cE);
    k_mix_experts<<<gb(cB * cF), 256, 0, stream>>>(gf, feBEF, fout, cE, 0, dm);

    // transformer block vmapped over 18 experts
    k_pack3<<<gb(cB * 3 * cF), 256, 0, stream>>>(tout, iout, fout, tokens);
    k_bcast_tokens<<<gb((long)cE * 24 * cF), 256, 0, stream>>>(tokens, xbuf);
    k_layernorm_expert<<<gb(cE * 24), 256, 0, stream>>>(xbuf, IN[48], IN[49], hbuf, 24, dm);
    k_expert_linear<<<gb((long)cE * 24 * 960), 256, 0, stream>>>(
        hbuf, (long)24 * cF, IN[50], IN[51], qkvb, 24, cF, 960, 0, dm, 0);
    k_attention<<<gb(cE * cB * 8), 256, 0, stream>>>(qkvb, attb);
    k_expert_linear<<<gb((long)cE * 24 * cF), 256, 0, stream>>>(
        attb, (long)24 * cF, IN[52], IN[53], projb, 24, cF, cF, 0, dm, 0);
    k_residual_add<<<gb((long)cE * 24 * cF), 256, 0, stream>>>(xbuf, projb, (long)cE * 24 * cF);
    k_layernorm_expert<<<gb(cE * 24), 256, 0, stream>>>(xbuf, IN[54], IN[55], hbuf, 24, dm);
    k_expert_linear<<<gb((long)cE * 24 * 1280), 256, 0, stream>>>(
        hbuf, (long)24 * cF, IN[56], IN[57], mlpb, 24, cF, 1280, 3, dm, 0);
    k_expert_linear<<<gb((long)cE * 24 * cF), 256, 0, stream>>>(
        mlpb, (long)24 * 1280, IN[58], IN[59], projb, 24, 1280, cF, 0, dm, 0);
    k_residual_add<<<gb((long)cE * 24 * cF), 256, 0, stream>>>(xbuf, projb, (long)cE * 24 * cF);
    k_mean_tokens<<<gb(cE * cB * cF), 256, 0, stream>>>(xbuf, blk);

    // noise gate over block outputs
    k_dense_linear<<<gb(cB * cF), 256, 0, stream>>>(fout, IN[38] + (long)dm * cF * cF,
                                                    IN[39] + dm * cF, nh1, cB, cF, cF, 1);
    k_dense_linear<<<gb(cB * 160), 256, 0, stream>>>(nh1, IN[40] + (long)dm * cF * 160,
                                                     IN[41] + dm * 160, nh2, cB, cF, 160, 1);
    k_dense_linear<<<gb(cB * cE), 256, 0, stream>>>(nh2, IN[42] + (long)dm * 160 * cE,
                                                    IN[43] + dm * cE, gn, cB, 160, cE, 0);
    k_softmax_rows<<<1, 32, 0, stream>>>(gn, cB, cE);
    k_mix_ebf<<<gb(cB * cF), 256, 0, stream>>>(gn, blk, finl);

    // Sinkhorn-OT prototype dissonance
    k_proto_scores<<<1, 96, 0, stream>>>(tout, IN[64], st);
    k_proto_scores<<<1, 96, 0, stream>>>(iout, IN[65], si);
    k_sinkhorn<<<1, 32, 0, stream>>>(st, si, Tm);
    k_dense_linear<<<gb(cB * 128), 256, 0, stream>>>(Tm, IN[66], IN[67], dh, cB, 144, 128, 3);
    k_dense_linear<<<gb(cB * cF), 256, 0, stream>>>(dh, IN[68], IN[69], dis, cB, 128, cF, 0);

    // fusion + classifier head
    k_concat2<<<gb(cB * 2 * cF), 256, 0, stream>>>(finl, dis, cat);
    k_dense_linear<<<gb(cB * cF), 256, 0, stream>>>(cat, IN[70], IN[71], fh, cB, 2 * cF, cF, 1);
    k_dense_linear<<<gb(cB * cF), 256, 0, stream>>>(fh, IN[72], IN[73], fused, cB, cF, cF, 0);
    k_dense_linear<<<gb(cB * 384), 256, 0, stream>>>(fused, IN[60] + (long)dm * cF * 384,
                                                     IN[61] + dm * 384, ch, cB, cF, 384, 2);
    k_dense_linear<<<gb(cB), 256, 0, stream>>>(ch, IN[62] + (long)dm * 384,
                                               IN[63] + dm, lgt, cB, 384, 1, 0);
    k_sigmoid_out<<<1, 32, 0, stream>>>(lgt, out, dm);
  }
}
